// ColBERTLoss_52948356825229
// MI455X (gfx1250) — matrix operand order repaired and checked by reading, not compiled, not run on hardware
//
#include <hip/hip_runtime.h>
#include <hip/hip_bf16.h>

typedef __attribute__((ext_vector_type(16))) __bf16 v16bf;
typedef __attribute__((ext_vector_type(8)))  float  v8f;

namespace {
constexpr int kB    = 64;   // batch
constexpr int kQ    = 32;   // query tokens
constexpr int kD    = 256;  // doc tokens
constexpr int kDim  = 128;  // embedding dim
constexpr float kInvTemp = 20.0f;  // 1 / 0.05
}

__device__ __forceinline__ unsigned short f2bf_rn(float f) {
    unsigned u = __float_as_uint(f);
    unsigned r = (u + 0x7FFFu + ((u >> 16) & 1u)) >> 16;
    return (unsigned short)r;
}
__device__ __forceinline__ float bf2f(unsigned short h) {
    return __uint_as_float(((unsigned)h) << 16);
}

// ---------------------------------------------------------------------------
// Kernel 1: split fp32 -> (hi, lo) bf16 planes.  x ~= hi + lo with lo the
// round-to-nearest residual, giving ~2^-16 relative error on products.
// ---------------------------------------------------------------------------
__global__ __launch_bounds__(256) void colbert_split_kernel(
    const float* __restrict__ q, const float* __restrict__ p,
    unsigned short* __restrict__ Qhi, unsigned short* __restrict__ Qlo,
    unsigned short* __restrict__ Phi, unsigned short* __restrict__ Plo)
{
    constexpr int NQ = kB * kQ * kDim / 4;   // float4 groups in Q
    constexpr int NP = kB * kD * kDim / 4;   // float4 groups in P
    int idx = blockIdx.x * blockDim.x + threadIdx.x;
    const float4* src;
    unsigned short* hi;
    unsigned short* lo;
    int j;
    if (idx < NQ)            { src = (const float4*)q; hi = Qhi; lo = Qlo; j = idx; }
    else if (idx < NQ + NP)  { src = (const float4*)p; hi = Phi; lo = Plo; j = idx - NQ; }
    else return;

    float4 f = src[j];
    ushort4 h, l;
    h.x = f2bf_rn(f.x); l.x = f2bf_rn(f.x - bf2f(h.x));
    h.y = f2bf_rn(f.y); l.y = f2bf_rn(f.y - bf2f(h.y));
    h.z = f2bf_rn(f.z); l.z = f2bf_rn(f.z - bf2f(h.z));
    h.w = f2bf_rn(f.w); l.w = f2bf_rn(f.w - bf2f(h.w));
    ((ushort4*)hi)[j] = h;
    ((ushort4*)lo)[j] = l;
}

// ---------------------------------------------------------------------------
// Kernel 2: one block per (a,b) pair.  8 waves compute the 32x256 score tile
// with v_wmma_f32_16x16x32_bf16 (3-product split for ~fp32 accuracy), fuse
// pos-mask + row-max + query-mask + sum in registers/LDS -> sim[a][b].
// All fragment addresses are one per-lane base pointer per plane plus
// compile-time-constant offsets (fit the signed 24-bit IOFFSET), so the
// unrolled loop body is pure global_load_b128+imm / v_wmma.
// ---------------------------------------------------------------------------
union FragU { uint4 u[2]; v16bf v; };

__global__ __launch_bounds__(256) void colbert_maxsim_kernel(
    const unsigned short* __restrict__ Qhi, const unsigned short* __restrict__ Qlo,
    const unsigned short* __restrict__ Phi, const unsigned short* __restrict__ Plo,
    const unsigned char* __restrict__ qmask, const unsigned char* __restrict__ pmask,
    float* __restrict__ sim)
{
    const int a    = blockIdx.y;
    const int b    = blockIdx.x;
    const int tid  = threadIdx.x;
    const int wave = tid >> 5;
    const int lane = tid & 31;
    const int half = lane >> 4;   // which 16-lane half
    const int l16  = lane & 15;
    const int mtile = wave & 1;   // row tile: 0 -> rows 0..15, 1 -> rows 16..31
    const int ngrp  = wave >> 1;  // 4 consecutive N tiles per wave

    // --- A fragments: row (mtile*16 + l16) of query batch a, all 4 K-steps.
    // 16-bit A layout: lane row m holds K chunks {k0..k0+7, k0+16..k0+23},
    // upper lane half shifted by +8; fold the +half*8 into the base.
    const size_t qoff = (size_t)(a * kQ + mtile * 16 + l16) * kDim + half * 8;
    const unsigned short* qh = Qhi + qoff;
    const unsigned short* ql = Qlo + qoff;
    v16bf Ahi[4], Alo[4];
#pragma unroll
    for (int kk = 0; kk < 4; ++kk) {
        FragU fh, fl;
        fh.u[0] = *(const uint4*)(qh + kk * 32);
        fh.u[1] = *(const uint4*)(qh + kk * 32 + 16);
        fl.u[0] = *(const uint4*)(ql + kk * 32);
        fl.u[1] = *(const uint4*)(ql + kk * 32 + 16);
        Ahi[kk] = fh.v;
        Alo[kk] = fl.v;
    }

    // --- B bases: lane owns doc token p(t) = ngrp*64 + t*16 + l16 of batch b.
    // 16-bit B layout: lane holds K chunks {k0..k0+7, k0+16..k0+23},
    // upper lane half shifted by +8; fold the +half*8 into the base.  Tile stride
    // = 16 rows * 128 = 2048 ushorts, K-step stride = 32 ushorts: all loads
    // become base + constant IOFFSET.
    const size_t poff = (size_t)(b * kD + ngrp * 64 + l16) * kDim + half * 8;
    const unsigned short* phB = Phi + poff;
    const unsigned short* plB = Plo + poff;
    const unsigned char*  pmB = pmask + b * kD + ngrp * 64 + l16;

    const float NEG = -__builtin_inff();
    // Per-lane running max over the 4 columns this lane owns (one per N-tile).
    float rmax[8];
#pragma unroll
    for (int v = 0; v < 8; ++v) rmax[v] = NEG;

#pragma unroll
    for (int t = 0; t < 4; ++t) {
        v8f acc = {};
#pragma unroll
        for (int kk = 0; kk < 4; ++kk) {
            const int o = t * 2048 + kk * 32;   // constant after unroll
            FragU bh, bl;
            bh.u[0] = *(const uint4*)(phB + o);
            bh.u[1] = *(const uint4*)(phB + o + 16);
            bl.u[0] = *(const uint4*)(plB + o);
            bl.u[1] = *(const uint4*)(plB + o + 16);
            // (hi+lo)x(hi+lo) ~= hi*hi + hi*lo + lo*hi  (fp32 accumulate)
            acc = __builtin_amdgcn_wmma_f32_16x16x32_bf16(false, Ahi[kk], false, bh.v,
                                                          (short)0, acc, false, false);
            acc = __builtin_amdgcn_wmma_f32_16x16x32_bf16(false, Ahi[kk], false, bl.v,
                                                          (short)0, acc, false, false);
            acc = __builtin_amdgcn_wmma_f32_16x16x32_bf16(false, Alo[kk], false, bh.v,
                                                          (short)0, acc, false, false);
        }
        // C layout: lane -> column N = l16, VGPR v -> row M = v + 8*half.
        const bool dvalid = pmB[t * 16] != 0;
#pragma unroll
        for (int v = 0; v < 8; ++v)
            rmax[v] = fmaxf(rmax[v], dvalid ? acc[v] : NEG);
    }

    // Deferred cross-lane max: one 4-step xor reduction per accumulator row
    // (16-lane halves hold independent rows, xor offsets 1..8 stay in-half).
#pragma unroll
    for (int v = 0; v < 8; ++v) {
        float x = rmax[v];
#pragma unroll
        for (int off = 1; off < 16; off <<= 1)
            x = fmaxf(x, __shfl_xor(x, off, 32));
        rmax[v] = x;
    }

    // Cross-wave row-max: wave w covers rows [mtile*16, mtile*16+16).
    __shared__ float smax[8][32];
    if (l16 == 0) {
#pragma unroll
        for (int v = 0; v < 8; ++v)
            smax[wave][mtile * 16 + half * 8 + v] = rmax[v];
    }
    __syncthreads();

    if (tid < 32) {  // one full wave: row r = tid
        const int r   = tid;
        const int par = r >> 4;  // which mtile owns this row
        float m = fmaxf(fmaxf(smax[par][r],     smax[par + 2][r]),
                        fmaxf(smax[par + 4][r], smax[par + 6][r]));
        float val = (qmask[a * kQ + r] != 0) ? m : 0.0f;
#pragma unroll
        for (int off = 1; off < 32; off <<= 1)
            val += __shfl_xor(val, off, 32);
        if (r == 0) sim[a * kB + b] = val * kInvTemp;
    }
}

// ---------------------------------------------------------------------------
// Kernel 3: cross-entropy with in-batch negatives over sim (64x64), diagonal
// labels.  64 threads, one row each; tiny.
// ---------------------------------------------------------------------------
__global__ __launch_bounds__(64) void colbert_loss_kernel(
    const float* __restrict__ sim, float* __restrict__ out)
{
    __shared__ float red[64];
    const int i = threadIdx.x;
    const float* row = sim + i * kB;
    float m = -__builtin_inff();
    for (int j = 0; j < kB; ++j) m = fmaxf(m, row[j]);
    float s = 0.0f;
    for (int j = 0; j < kB; ++j) s += expf(row[j] - m);
    const float lse = m + logf(s);
    red[i] = row[i] - lse;  // logp[i,i]
    __syncthreads();
    for (int off = 32; off > 0; off >>= 1) {
        if (i < off) red[i] += red[i + off];
        __syncthreads();
    }
    if (i == 0) out[0] = -red[0] * (1.0f / (float)kB);
}

// ---------------------------------------------------------------------------
extern "C" void kernel_launch(void* const* d_in, const int* in_sizes, int n_in,
                              void* d_out, int out_size, void* d_ws, size_t ws_size,
                              hipStream_t stream)
{
    const float*         q     = (const float*)d_in[0];
    const float*         p     = (const float*)d_in[1];
    const unsigned char* qmask = (const unsigned char*)d_in[2];  // bool mask
    const unsigned char* pmask = (const unsigned char*)d_in[3];  // bool mask
    float*               out   = (float*)d_out;

    // Workspace layout (~9.02 MB): bf16 hi/lo planes + sim matrix.
    unsigned short* Qhi = (unsigned short*)d_ws;
    unsigned short* Qlo = Qhi + (size_t)kB * kQ * kDim;
    unsigned short* Phi = Qlo + (size_t)kB * kQ * kDim;
    unsigned short* Plo = Phi + (size_t)kB * kD * kDim;
    float*          sim = (float*)(Plo + (size_t)kB * kD * kDim);

    const int ngroups = (kB * kQ * kDim + kB * kD * kDim) / 4;  // 589824
    colbert_split_kernel<<<(ngroups + 255) / 256, 256, 0, stream>>>(
        q, p, Qhi, Qlo, Phi, Plo);

    colbert_maxsim_kernel<<<dim3(kB, kB), 256, 0, stream>>>(
        Qhi, Qlo, Phi, Plo, qmask, pmask, sim);

    colbert_loss_kernel<<<1, 64, 0, stream>>>(sim, out);
}
